// LSMultiheadAttentionLayer_38912403701862
// MI455X (gfx1250) — matrix operand order repaired and checked by reading, not compiled, not run on hardware
//
#include <hip/hip_runtime.h>

// ---------------------------------------------------------------------------
// LSMultiheadAttentionLayer for MI455X (gfx1250), wave32 + WMMA f16 16x16x32
// B=4, S=2048, HS=1024, NH=16, HD=64
// GEMMs use double-buffered LDS fed by global_load_async_to_lds_b128
// (ASYNCcnt-tracked DMA, no VGPR round trip), WMMA f16 w/ f32 accumulation.
// Workspace usage: ~88 MB of f16 staging buffers carved from d_ws.
// ---------------------------------------------------------------------------

#define HS    1024
#define NH    16
#define HD    64
#define BDIM  4
#define SDIM  2048
#define MROWS (BDIM * SDIM)   // 8192
#define NEG_INF  (-1e8f)
#define LN_EPS   (1e-5f)

typedef __attribute__((ext_vector_type(16))) _Float16 v16h;
typedef __attribute__((ext_vector_type(8)))  float    v8f;
typedef _Float16 h16;

union AF {                 // 16-half fragment (A or B operand), 8 VGPRs
  v16h      v;
  h16       h[16];
  unsigned  u[8];
};
union CF {                 // 8-float accumulator fragment
  v8f   v;
  float f[8];
};

static __device__ __forceinline__ v8f wmma_f16(v16h a, v16h b, v8f c) {
  return __builtin_amdgcn_wmma_f32_16x16x32_f16(
      /*neg_a=*/false, a, /*neg_b=*/false, b,
      /*c_mod=*/(short)0, c, /*reuse_a=*/false, /*reuse_b=*/false);
}

// Async DMA: 16 bytes global -> LDS, imm offset added to both addresses, ASYNCcnt-tracked.
static __device__ __forceinline__ void async_b128(unsigned lds_off,
                                                  const void* gaddr) {
  asm volatile("global_load_async_to_lds_b128 %0, %1, off"
               :: "v"(lds_off), "v"(gaddr) : "memory");
}
static __device__ __forceinline__ void async_b128_off16(unsigned lds_off,
                                                        const void* gaddr) {
  asm volatile("global_load_async_to_lds_b128 %0, %1, off offset:16"
               :: "v"(lds_off), "v"(gaddr) : "memory");
}
static __device__ __forceinline__ void wait_async0() {
  asm volatile("s_wait_asynccnt 0x0" ::: "memory");
}
static __device__ __forceinline__ unsigned lds_off32(const void* p) {
  return (unsigned)(size_t)p;   // generic LDS addr: low 32 bits = LDS offset
}

// ---------------------------------------------------------------------------
// Kernel 0: f32 -> f16 conversion (weights)
// ---------------------------------------------------------------------------
__global__ void __launch_bounds__(256)
cvt_f32_to_f16(const float* __restrict__ src, h16* __restrict__ dst, int n) {
  int i = blockIdx.x * 256 + threadIdx.x;
  if (i < n) dst[i] = (h16)src[i];
}

// ---------------------------------------------------------------------------
// Kernel 1: LayerNorm -> f16   (one block of 256 threads per row of 1024)
// ---------------------------------------------------------------------------
__global__ void __launch_bounds__(256)
ln_f16_kernel(const float* __restrict__ x, const float* __restrict__ g,
              const float* __restrict__ be, h16* __restrict__ y) {
  const int row = blockIdx.x;
  const int t   = threadIdx.x;
  const float* xr = x + (size_t)row * HS;
  float4 v = reinterpret_cast<const float4*>(xr)[t];
  float s  = v.x + v.y + v.z + v.w;
  float ss = v.x * v.x + v.y * v.y + v.z * v.z + v.w * v.w;
  #pragma unroll
  for (int m = 1; m < 32; m <<= 1) {
    s  += __shfl_xor(s,  m, 32);
    ss += __shfl_xor(ss, m, 32);
  }
  __shared__ float as[8], aq[8];
  const int w = t >> 5, l = t & 31;
  if (l == 0) { as[w] = s; aq[w] = ss; }
  __syncthreads();
  float S = 0.f, Q = 0.f;
  #pragma unroll
  for (int i = 0; i < 8; ++i) { S += as[i]; Q += aq[i]; }
  const float mu  = S * (1.0f / HS);
  const float var = Q * (1.0f / HS) - mu * mu;
  const float rs  = rsqrtf(var + LN_EPS);
  h16* yr = y + (size_t)row * HS;
  const int c = t * 4;
  yr[c + 0] = (h16)((v.x - mu) * rs * g[c + 0] + be[c + 0]);
  yr[c + 1] = (h16)((v.y - mu) * rs * g[c + 1] + be[c + 1]);
  yr[c + 2] = (h16)((v.z - mu) * rs * g[c + 2] + be[c + 2]);
  yr[c + 3] = (h16)((v.w - mu) * rs * g[c + 3] + be[c + 3]);
}

// ---------------------------------------------------------------------------
// Kernels 2 & 4: GEMM  C[M,N] = A[M,K] * W[N,K]^T + bias  (f16 in, f32 acc)
// Block tile 128(M) x 64(N), K-step 32, 8 waves (4x2), 32x32 per wave.
// Double-buffered LDS, filled by async global->LDS DMA; DMA of tile t+1
// overlaps WMMA compute of tile t.
// EPI==0: QKV epilogue (scatter f16 into q/k/v [B,NH,S,HD])
// EPI==1: out-proj epilogue (f32 out = acc + bias + residual)
// ---------------------------------------------------------------------------
template <int EPI>
__global__ void __launch_bounds__(256)
gemm_wmma_kernel(const h16* __restrict__ A, const h16* __restrict__ W,
                 const float* __restrict__ bias,
                 h16* __restrict__ q16, h16* __restrict__ k16, h16* __restrict__ v16,
                 const float* __restrict__ residual, float* __restrict__ outF,
                 int Kdim) {
  __shared__ alignas(16) h16 sA[2][128][40];  // 128 rows x 32 K (pad 40 halfs)
  __shared__ alignas(16) h16 sB[2][64][40];   // 64 N-rows x 32 K

  const int tid  = threadIdx.x;
  const int wave = tid >> 5, lane = tid & 31;
  const int wm = wave >> 1, wn = wave & 1;
  const int lr = lane & 15, hi = lane >> 4;
  const int m0 = blockIdx.y * 128;
  const int n0 = blockIdx.x * 64;

  CF acc[2][2];
  #pragma unroll
  for (int mi = 0; mi < 2; ++mi)
    #pragma unroll
    for (int ni = 0; ni < 2; ++ni)
      #pragma unroll
      for (int r = 0; r < 8; ++r) acc[mi][ni].f[r] = 0.f;

  const int arow = tid >> 1, aoff = (tid & 1) * 16;  // 16 halfs -> 2 x b128
  const int brow = tid >> 2, boff = (tid & 3) * 8;   // 8 halfs  -> 1 x b128
  const h16* gA = A + (size_t)(m0 + arow) * Kdim + aoff;
  const h16* gB = W + (size_t)(n0 + brow) * Kdim + boff;

  const int nTiles = Kdim >> 5;  // K-step 32

  // Prologue: DMA tile 0 into buffer 0.
  {
    async_b128(lds_off32(&sA[0][arow][aoff]), gA);
    async_b128_off16(lds_off32(&sA[0][arow][aoff]), gA);
    async_b128(lds_off32(&sB[0][brow][boff]), gB);
  }

  for (int t = 0; t < nTiles; ++t) {
    const int cur = t & 1;
    wait_async0();     // my DMA for tile t has landed in LDS
    __syncthreads();   // everyone's tile-t DMA landed; buf cur^1 free to fill

    if (t + 1 < nTiles) {
      const h16* gA1 = gA + (size_t)(t + 1) * 32;
      const h16* gB1 = gB + (size_t)(t + 1) * 32;
      async_b128(lds_off32(&sA[cur ^ 1][arow][aoff]), gA1);
      async_b128_off16(lds_off32(&sA[cur ^ 1][arow][aoff]), gA1);
      async_b128(lds_off32(&sB[cur ^ 1][brow][boff]), gB1);
    }

    AF aFr[2], bFr[2];
    #pragma unroll
    for (int mi = 0; mi < 2; ++mi) {
      const unsigned* pu = reinterpret_cast<const unsigned*>(
          &sA[cur][wm * 32 + mi * 16 + lr][0]);
      #pragma unroll
      for (int i = 0; i < 8; ++i) {
        int kk = (i >> 2) * 16 + (i & 3) * 2 + hi * 8;
        aFr[mi].u[i] = pu[kk >> 1];
      }
    }
    #pragma unroll
    for (int ni = 0; ni < 2; ++ni) {
      const h16* pr = &sB[cur][wn * 32 + ni * 16 + lr][hi * 8];
      *reinterpret_cast<float4*>(&bFr[ni].h[0]) = *reinterpret_cast<const float4*>(pr);
      *reinterpret_cast<float4*>(&bFr[ni].h[8]) = *reinterpret_cast<const float4*>(pr + 16);
    }
    #pragma unroll
    for (int mi = 0; mi < 2; ++mi)
      #pragma unroll
      for (int ni = 0; ni < 2; ++ni)
        acc[mi][ni].v = wmma_f16(aFr[mi].v, bFr[ni].v, acc[mi][ni].v);
  }

  // Epilogue. C layout: VGPR r -> row (hi*8 + r), column = lane&15.
  #pragma unroll
  for (int mi = 0; mi < 2; ++mi) {
    #pragma unroll
    for (int ni = 0; ni < 2; ++ni) {
      const int col = n0 + wn * 32 + ni * 16 + lr;
      const float bv = bias[col];
      #pragma unroll
      for (int r = 0; r < 8; ++r) {
        const int row = m0 + wm * 32 + mi * 16 + hi * 8 + r;
        const float val = acc[mi][ni].f[r] + bv;
        if (EPI == 0) {
          const int which = col >> 10;          // 0:q 1:k 2:v
          const int rem   = col & 1023;
          const int hh    = rem >> 6;           // head
          const int dd    = rem & 63;           // dim
          const int bb    = row >> 11;          // row / 2048
          const int ss    = row & 2047;
          h16* dst = (which == 0) ? q16 : (which == 1) ? k16 : v16;
          dst[(((size_t)(bb * NH + hh)) * SDIM + ss) * HD + dd] = (h16)val;
        } else {
          const size_t idx = (size_t)row * HS + col;
          outF[idx] = val + residual[idx];
        }
      }
    }
  }
}

// ---------------------------------------------------------------------------
// Kernel 3: flash attention. Block = 128 threads (4 waves), 64 q-rows per
// block (16 per wave). Streams K/V in 64-key tiles through LDS, online
// softmax in f32, P@V via WMMA with P staged f16 in LDS.
// ---------------------------------------------------------------------------
__global__ void __launch_bounds__(128)
attn_kernel(const h16* __restrict__ q16, const h16* __restrict__ k16,
            const h16* __restrict__ v16, const int* __restrict__ mask,
            h16* __restrict__ ctx) {
  __shared__ alignas(16) h16 sK[64][72];    // [key][hd]
  __shared__ alignas(16) h16 sVt[64][72];   // [hd][key]  (transposed)
  __shared__ alignas(16) h16 sP[64][72];    // [q-row][key], wave-local stripes

  const int tid = threadIdx.x, wave = tid >> 5, lane = tid & 31;
  const int lr = lane & 15, hi = lane >> 4;
  const int bh = blockIdx.y;
  const int b  = bh >> 4;       // / NH
  const int h  = bh & 15;
  const int q0 = blockIdx.x * 64;
  const float scale = 0.125f;   // 1/sqrt(64)

  // Q A-fragments (16 rows x 64 K, two K=32 chunks), loaded once.
  AF aQ[2];
  {
    const unsigned* qr = reinterpret_cast<const unsigned*>(
        q16 + ((size_t)bh * SDIM + q0 + wave * 16 + lr) * HD);
    #pragma unroll
    for (int c = 0; c < 2; ++c)
      #pragma unroll
      for (int i = 0; i < 8; ++i) {
        int kk = c * 32 + (i >> 2) * 16 + (i & 3) * 2 + hi * 8;
        aQ[c].u[i] = qr[kk >> 1];
      }
  }

  CF o[4];
  float mrow[8], lrow[8];
  #pragma unroll
  for (int t4 = 0; t4 < 4; ++t4)
    #pragma unroll
    for (int r = 0; r < 8; ++r) o[t4].f[r] = 0.f;
  #pragma unroll
  for (int r = 0; r < 8; ++r) { mrow[r] = -3.0e38f; lrow[r] = 0.f; }

  const int ldrow = tid >> 1, ldoff = (tid & 1) * 32;  // 32 halfs per thread

  for (int kt = 0; kt < SDIM / 64; ++kt) {
    const float4* gk = reinterpret_cast<const float4*>(
        k16 + ((size_t)bh * SDIM + kt * 64 + ldrow) * HD + ldoff);
    const float4* gv = reinterpret_cast<const float4*>(
        v16 + ((size_t)bh * SDIM + kt * 64 + ldrow) * HD + ldoff);
    float4 ka = gk[0], kb2 = gk[1], kc = gk[2], kd = gk[3];
    float4 va = gv[0], vb = gv[1], vc = gv[2], vd = gv[3];

    __syncthreads();
    {
      float4* dk = reinterpret_cast<float4*>(&sK[ldrow][ldoff]);
      dk[0] = ka; dk[1] = kb2; dk[2] = kc; dk[3] = kd;
      // V stored transposed: sVt[hd][key]
      h16 vh[32];
      *reinterpret_cast<float4*>(&vh[0])  = va;
      *reinterpret_cast<float4*>(&vh[8])  = vb;
      *reinterpret_cast<float4*>(&vh[16]) = vc;
      *reinterpret_cast<float4*>(&vh[24]) = vd;
      #pragma unroll
      for (int j = 0; j < 32; ++j) sVt[ldoff + j][ldrow] = vh[j];
    }
    __syncthreads();

    // ---- scores: S = Q * K^T, 4 key sub-tiles of 16 ----
    CF sc[4];
    #pragma unroll
    for (int nt = 0; nt < 4; ++nt) {
      #pragma unroll
      for (int r = 0; r < 8; ++r) sc[nt].f[r] = 0.f;
      #pragma unroll
      for (int c = 0; c < 2; ++c) {
        AF bK;
        const h16* pr = &sK[nt * 16 + lr][c * 32 + hi * 8];
        *reinterpret_cast<float4*>(&bK.h[0]) = *reinterpret_cast<const float4*>(pr);
        *reinterpret_cast<float4*>(&bK.h[8]) = *reinterpret_cast<const float4*>(pr + 16);
        sc[nt].v = wmma_f16(aQ[c].v, bK.v, sc[nt].v);
      }
      const float mv =
          (float)mask[b * SDIM + kt * 64 + nt * 16 + lr] * NEG_INF;
      #pragma unroll
      for (int r = 0; r < 8; ++r) sc[nt].f[r] = sc[nt].f[r] * scale + mv;
    }

    // ---- online softmax (each lane owns rows hi*8 + r) ----
    #pragma unroll
    for (int r = 0; r < 8; ++r) {
      float m = fmaxf(fmaxf(sc[0].f[r], sc[1].f[r]),
                      fmaxf(sc[2].f[r], sc[3].f[r]));
      #pragma unroll
      for (int d = 1; d < 16; d <<= 1) m = fmaxf(m, __shfl_xor(m, d, 32));
      const float mnew = fmaxf(mrow[r], m);
      const float corr = __expf(mrow[r] - mnew);
      mrow[r] = mnew;
      float rs = 0.f;
      #pragma unroll
      for (int nt = 0; nt < 4; ++nt) {
        float p = __expf(sc[nt].f[r] - mnew);
        sc[nt].f[r] = p;
        rs += p;
      }
      #pragma unroll
      for (int d = 1; d < 16; d <<= 1) rs += __shfl_xor(rs, d, 32);
      lrow[r] = lrow[r] * corr + rs;
      #pragma unroll
      for (int t4 = 0; t4 < 4; ++t4) o[t4].f[r] *= corr;
    }

    // ---- stage P (f16) into wave-local LDS stripe, rebuild as A frags ----
    #pragma unroll
    for (int nt = 0; nt < 4; ++nt)
      #pragma unroll
      for (int r = 0; r < 8; ++r)
        sP[wave * 16 + hi * 8 + r][nt * 16 + lr] = (h16)sc[nt].f[r];

    AF aP[2];
    {
      const unsigned* pu = reinterpret_cast<const unsigned*>(
          &sP[wave * 16 + lr][0]);
      #pragma unroll
      for (int c = 0; c < 2; ++c)
        #pragma unroll
        for (int i = 0; i < 8; ++i) {
          int kk = c * 32 + (i >> 2) * 16 + (i & 3) * 2 + hi * 8;
          aP[c].u[i] = pu[kk >> 1];
        }
    }

    // ---- O += P * V ----
    #pragma unroll
    for (int t4 = 0; t4 < 4; ++t4) {
      #pragma unroll
      for (int c = 0; c < 2; ++c) {
        AF bV;
        const h16* pr = &sVt[t4 * 16 + lr][c * 32 + hi * 8];
        *reinterpret_cast<float4*>(&bV.h[0]) = *reinterpret_cast<const float4*>(pr);
        *reinterpret_cast<float4*>(&bV.h[8]) = *reinterpret_cast<const float4*>(pr + 16);
        o[t4].v = wmma_f16(aP[c].v, bV.v, o[t4].v);
      }
    }
  }

  // ---- normalize + write ctx f16 in [B, S, HS] layout ----
  #pragma unroll
  for (int t4 = 0; t4 < 4; ++t4) {
    const int col = h * HD + t4 * 16 + lr;
    #pragma unroll
    for (int r = 0; r < 8; ++r) {
      const int row = q0 + wave * 16 + hi * 8 + r;
      const float inv = 1.0f / lrow[r];
      ctx[((size_t)b * SDIM + row) * HS + col] = (h16)(o[t4].f[r] * inv);
    }
  }
}

// ---------------------------------------------------------------------------
// Launcher
// ---------------------------------------------------------------------------
extern "C" void kernel_launch(void* const* d_in, const int* in_sizes, int n_in,
                              void* d_out, int out_size, void* d_ws, size_t ws_size,
                              hipStream_t stream) {
  const float* hs    = (const float*)d_in[0];
  const int*   mask  = (const int*)  d_in[1];
  const float* qkv_w = (const float*)d_in[2];
  const float* qkv_b = (const float*)d_in[3];
  const float* out_w = (const float*)d_in[4];
  const float* out_b = (const float*)d_in[5];
  const float* ln_g  = (const float*)d_in[6];
  const float* ln_b  = (const float*)d_in[7];
  float* out = (float*)d_out;

  // Carve f16 staging from workspace (~88 MB total).
  h16* p = (h16*)d_ws;
  h16* x16   = p; p += (size_t)MROWS * HS;          // LN output
  h16* qw16  = p; p += (size_t)3 * HS * HS;         // qkv weights
  h16* ow16  = p; p += (size_t)HS * HS;             // out weights
  h16* q16   = p; p += (size_t)BDIM * NH * SDIM * HD;
  h16* k16   = p; p += (size_t)BDIM * NH * SDIM * HD;
  h16* v16   = p; p += (size_t)BDIM * NH * SDIM * HD;
  h16* ctx16 = p; p += (size_t)MROWS * HS;

  cvt_f32_to_f16<<<(3 * HS * HS) / 256, 256, 0, stream>>>(qkv_w, qw16, 3 * HS * HS);
  cvt_f32_to_f16<<<(HS * HS) / 256, 256, 0, stream>>>(out_w, ow16, HS * HS);

  ln_f16_kernel<<<MROWS, 256, 0, stream>>>(hs, ln_g, ln_b, x16);

  // QKV: [8192,1024] x [3072,1024]^T
  gemm_wmma_kernel<0><<<dim3(3 * HS / 64, MROWS / 128), 256, 0, stream>>>(
      x16, qw16, qkv_b, q16, k16, v16, nullptr, nullptr, HS);

  // Attention: grid (S/64 q-tiles, B*NH)
  attn_kernel<<<dim3(SDIM / 64, BDIM * NH), 128, 0, stream>>>(
      q16, k16, v16, mask, ctx16);

  // Out-proj: [8192,1024] x [1024,1024]^T + bias + residual
  gemm_wmma_kernel<1><<<dim3(HS / 64, MROWS / 128), 256, 0, stream>>>(
      ctx16, ow16, out_b, nullptr, nullptr, nullptr, hs, out, HS);
}
